// DynamicRNN_35923106464006
// MI455X (gfx1250) — matrix operand order repaired and checked by reading, not compiled, not run on hardware
//
#include <hip/hip_runtime.h>
#include <hip/hip_bf16.h>
#include <math.h>

// ---------------------------------------------------------------------------
// 2-layer LSTM (B=64, T=1024, D=64, H=256) + FC head, for MI455X (gfx1250).
//
//  * f16 WMMA (v_wmma_f32_16x16x32_f16) for all matmuls, fp32 accumulate.
//  * Persistent recurrence kernels: 4 blocks (16-batch-row tiles) x 16 waves.
//    Wave w owns hidden columns [16w,16w+16) for ALL FOUR gates -> LSTM
//    pointwise update entirely in registers; c-state lives in a v8f.
//  * Weights pre-packed fragment-major f16; streamed from L2 every step.
//    Base pointers laundered (SGPR) each iteration so LICM cannot hoist the
//    loop-invariant fragment loads and spill; indexing is SGPR-base +
//    32-bit offset so the backend selects GVS addressing (saddr + imm).
//  * Gate biases pre-broadcast into v8f and used as the first WMMA's
//    C-operand: zero pointwise bias cost on the serial critical path.
//  * Input GEMM fused into the recurrence (no 128MB xg buffer).
// ---------------------------------------------------------------------------

typedef __attribute__((ext_vector_type(16))) _Float16 v16h;
typedef __attribute__((ext_vector_type(8)))  _Float16 v8h;
typedef __attribute__((ext_vector_type(8)))  float    v8f;

// Global (device memory, address_space(1)) views for laundered weight loads.
typedef __attribute__((address_space(1))) const _Float16 gf16;
typedef __attribute__((address_space(1))) const v16h    gv16h;

#define LSTM_B  64
#define LSTM_T  1024
#define LSTM_D  64
#define LSTM_H  256
#define LSTM_G  1024   // 4*H

// ---------------------------------------------------------------------------
// Pack fp32 weight matrix W (1024 x K, row-major, gates-major) into f16
// B-fragment-major layout. Fragment f = nt*(K/32)+kt is 512 halves:
//   out[f*512 + L*16 + j] = W[nt*16 + (L&15)][kt*32 + (L>=16?8:0) + (j<8 ? j : j+8)]
// matching the documented 16-bit B (32x16) VGPR striping for wave32 WMMA.
// ---------------------------------------------------------------------------
__global__ void lstm_pack_w(const float* __restrict__ W,
                            _Float16* __restrict__ out, int K) {
  int tid = blockIdx.x * blockDim.x + threadIdx.x;
  int total = LSTM_G * K;
  if (tid >= total) return;
  int f = tid >> 9;          // /512
  int r = tid & 511;
  int L = r >> 4;
  int j = r & 15;
  int ktiles = K >> 5;
  int nt = f / ktiles;
  int kt = f - nt * ktiles;
  int n = nt * 16 + (L & 15);
  int k = kt * 32 + ((L >= 16) ? 8 : 0) + ((j < 8) ? j : j + 8);
  out[tid] = (_Float16)W[(size_t)n * K + k];
}

// A-fragment (16xK tile, 16-bit, wave32) loader from row-major LDS [16][RS]:
// lane L holds row m=L&15; halves cover K = kb..kb+7 and kb+16..kb+23,
// kb = kt*32 + (L>=16 ? 8 : 0)  (documented 16-bit A striping).
__device__ __forceinline__ v16h lstm_load_a(const _Float16* base, int RS,
                                            int kt, int lane) {
  int m  = lane & 15;
  int kb = kt * 32 + ((lane >= 16) ? 8 : 0);
  const _Float16* p0 = base + m * RS + kb;
  const _Float16* p1 = p0 + 16;
  v16h a;
#pragma unroll
  for (int j = 0; j < 8; ++j) { a[j] = p0[j]; a[j + 8] = p1[j]; }
  return a;
}

// Weight B-fragment load: uniform SGPR base (laundered), 32-bit element
// offset -> backend selects GLOBAL GVS addressing (saddr + voffset + imm).
__device__ __forceinline__ v16h lstm_load_b(uintptr_t base_u, int elem_off) {
  const gf16* bp = reinterpret_cast<const gf16*>(base_u);
  return *(const gv16h*)(bp + elem_off);
}

__device__ __forceinline__ float lstm_sigmoid(float x) {
  return 1.0f / (1.0f + __expf(-x));
}
__device__ __forceinline__ float lstm_tanh(float x) {
  float e = __expf(2.0f * x);
  return (e - 1.0f) / (e + 1.0f);
}

// ---------------------------------------------------------------------------
// Persistent recurrence kernel. LAYER==0: input = x (fp32, B,T,D), writes f16
// h1 trace. LAYER==1: input = h1 (f16, B,T,H), computes FC head at t=T-1.
// Grid: 4 blocks x 512 threads (16 waves).
// ---------------------------------------------------------------------------
template <int LAYER>
__global__ __launch_bounds__(512) void lstm_rec(
    const float* __restrict__ x,        // LAYER==0 only
    const _Float16* __restrict__ h_in,  // LAYER==1 only
    const _Float16* __restrict__ Wih_pk,
    const _Float16* __restrict__ Whh_pk,
    const float* __restrict__ b_ih, const float* __restrict__ b_hh,
    _Float16* __restrict__ h_out,       // LAYER==0 only
    const float* __restrict__ fc_w, const float* __restrict__ fc_b,
    float* __restrict__ out)            // LAYER==1 only
{
  constexpr int KIN = (LAYER == 0) ? LSTM_D : LSTM_H;
  constexpr int NKX = KIN / 32;    // k-tiles of the input GEMM (2 or 8)
  constexpr int NKH = LSTM_H / 32; // k-tiles of the hidden GEMM (8)

  __shared__ _Float16 h_lds[16 * LSTM_H];  // current hidden state tile
  __shared__ _Float16 xt_lds[16 * KIN];    // staged input rows for step t

  const int tid  = threadIdx.x;
  const int lane = tid & 31;
  const int w    = tid >> 5;               // wave id: hidden tile [16w,16w+16)
  const int m0   = blockIdx.x * 16;        // batch-row tile
  const int ncol = lane & 15;
  const int mhi  = (lane >= 16) ? 8 : 0;
  const int lane_eoff = lane * 16;         // lane's 16 halves within a frag

  // Per-gate bias, broadcast into v8f accumulator seeds (loop-invariant).
  v8f bacc[4];
#pragma unroll
  for (int g = 0; g < 4; ++g) {
    int idx = g * LSTM_H + w * 16 + ncol;
    float bv = b_ih[idx] + b_hh[idx];
#pragma unroll
    for (int r = 0; r < 8; ++r) bacc[g][r] = bv;
  }

  // h = 0
  for (int i = tid; i < 16 * LSTM_H; i += 512) h_lds[i] = (_Float16)0.0f;
  // c = 0 (v8f register tile: (m = r+mhi, n = 16w+ncol))
  v8f cst = {};
  __syncthreads();

  for (int t = 0; t < LSTM_T; ++t) {
    // Launder weight base pointers each iteration (SGPR-resident) so the
    // loop-invariant fragment loads are NOT hoisted out of the t-loop and
    // spilled to scratch; they must stream from L2 every step.
    uintptr_t wih_u = (uintptr_t)Wih_pk;
    uintptr_t whh_u = (uintptr_t)Whh_pk;
    asm volatile("" : "+s"(wih_u), "+s"(whh_u));

    // ---- stage x_t tile into LDS (f16), prefetch t+1 ----
    if (LAYER == 0) {
      for (int i = tid; i < 16 * LSTM_D; i += 512) {
        int m = i >> 6, d = i & 63;
        const float* p = &x[((size_t)(m0 + m) * LSTM_T + t) * LSTM_D + d];
        xt_lds[i] = (_Float16)(*p);
        if (t + 1 < LSTM_T) __builtin_prefetch(p + LSTM_D, 0, 3);
      }
    } else {
      for (int i = tid; i < 16 * LSTM_H; i += 512) {
        int m = i >> 8, k = i & 255;
        const _Float16* p = &h_in[((size_t)(m0 + m) * LSTM_T + t) * LSTM_H + k];
        xt_lds[i] = *p;
        if (t + 1 < LSTM_T) __builtin_prefetch(p + LSTM_H, 0, 3);
      }
    }
    __syncthreads();

    // ---- load A fragments (shared across this wave's 4 gate tiles) ----
    v16h ax[NKX];
#pragma unroll
    for (int kt = 0; kt < NKX; ++kt) ax[kt] = lstm_load_a(xt_lds, KIN, kt, lane);
    v16h ah[NKH];
#pragma unroll
    for (int kt = 0; kt < NKH; ++kt) ah[kt] = lstm_load_a(h_lds, LSTM_H, kt, lane);

    // ---- gates = x_t @ W_ih^T + h @ W_hh^T + bias  (WMMA, fp32 acc) ----
    v8f gf[4];
#pragma unroll
    for (int g = 0; g < 4; ++g) {
      int nt = g * 16 + w;   // gate tile: columns [g*256 + 16w, +16)
      v8f acc = bacc[g];     // bias seeds the accumulator (C-operand)
#pragma unroll
      for (int kt = 0; kt < NKX; ++kt) {
        v16h bq = lstm_load_b(wih_u, (nt * NKX + kt) * 512 + lane_eoff);
        acc = __builtin_amdgcn_wmma_f32_16x16x32_f16(false, ax[kt], false, bq,
                                                     (short)0, acc, false, false);
      }
#pragma unroll
      for (int kt = 0; kt < NKH; ++kt) {
        v16h bq = lstm_load_b(whh_u, (nt * NKH + kt) * 512 + lane_eoff);
        acc = __builtin_amdgcn_wmma_f32_16x16x32_f16(false, ah[kt], false, bq,
                                                     (short)0, acc, false, false);
      }
      gf[g] = acc;
    }

    // ---- pointwise LSTM cell update, all in registers ----
    float hn[8];
#pragma unroll
    for (int r = 0; r < 8; ++r) {
      float iv = lstm_sigmoid(gf[0][r]);
      float fv = lstm_sigmoid(gf[1][r]);
      float gv = lstm_tanh(gf[2][r]);
      float ov = lstm_sigmoid(gf[3][r]);
      float cn = fv * cst[r] + iv * gv;
      cst[r] = cn;
      hn[r] = ov * lstm_tanh(cn);
    }

    __syncthreads();  // all h_lds reads of this step are done

    // ---- publish new h to LDS ----
#pragma unroll
    for (int r = 0; r < 8; ++r) {
      int m = r + mhi;
      h_lds[m * LSTM_H + w * 16 + ncol] = (_Float16)hn[r];
    }

    // ---- layer 0: coalesced f16 activation-trace store (b128 per thread) --
    if (LAYER == 0) {
      __syncthreads();
      // 16*256 halves = 8KB; 512 threads x 16B each, coalesced per row.
      int m = tid >> 5;            // batch row 0..15
      int hseg = (tid & 31) * 8;   // 8 contiguous halves
      v8h v = *(const v8h*)(h_lds + m * LSTM_H + hseg);
      *(v8h*)(&h_out[((size_t)(m0 + m) * LSTM_T + t) * LSTM_H + hseg]) = v;
    }
    __syncthreads();
  }

  // ---- FC head: out[b] = h2_last[b,:] . fc_w + fc_b ----
  if (LAYER == 1) {
    if (tid < 16) {
      int m = tid;
      float s = 0.0f;
      for (int k = 0; k < LSTM_H; ++k)
        s += (float)h_lds[m * LSTM_H + k] * fc_w[k];
      out[m0 + m] = s + fc_b[0];
    }
  }
}

// ---------------------------------------------------------------------------
// Launcher. Inputs (setup_inputs order):
//  0:x 1:W_ih0 2:W_hh0 3:b_ih0 4:b_hh0 5:W_ih1 6:W_hh1 7:b_ih1 8:b_hh1
//  9:fc_w 10:fc_b.  Output: 64 floats.  Workspace: ~33.6 MB.
// ---------------------------------------------------------------------------
extern "C" void kernel_launch(void* const* d_in, const int* in_sizes, int n_in,
                              void* d_out, int out_size, void* d_ws, size_t ws_size,
                              hipStream_t stream) {
  const float* x     = (const float*)d_in[0];
  const float* W_ih0 = (const float*)d_in[1];
  const float* W_hh0 = (const float*)d_in[2];
  const float* b_ih0 = (const float*)d_in[3];
  const float* b_hh0 = (const float*)d_in[4];
  const float* W_ih1 = (const float*)d_in[5];
  const float* W_hh1 = (const float*)d_in[6];
  const float* b_ih1 = (const float*)d_in[7];
  const float* b_hh1 = (const float*)d_in[8];
  const float* fc_w  = (const float*)d_in[9];
  const float* fc_b  = (const float*)d_in[10];
  float* out = (float*)d_out;

  char* ws = (char*)d_ws;
  const size_t SZ_WIH0 = (size_t)LSTM_G * LSTM_D * sizeof(_Float16);  // 128 KB
  const size_t SZ_WHH  = (size_t)LSTM_G * LSTM_H * sizeof(_Float16);  // 512 KB
  _Float16* wih0_pk = (_Float16*)(ws);
  _Float16* whh0_pk = (_Float16*)(ws + SZ_WIH0);
  _Float16* wih1_pk = (_Float16*)(ws + SZ_WIH0 + SZ_WHH);
  _Float16* whh1_pk = (_Float16*)(ws + SZ_WIH0 + 2 * SZ_WHH);
  _Float16* h1      = (_Float16*)(ws + SZ_WIH0 + 3 * SZ_WHH);
  (void)ws_size; (void)in_sizes; (void)n_in; (void)out_size;

  // Pack weights to f16 fragment-major layout.
  {
    int n0 = LSTM_G * LSTM_D;
    lstm_pack_w<<<(n0 + 255) / 256, 256, 0, stream>>>(W_ih0, wih0_pk, LSTM_D);
    int n1 = LSTM_G * LSTM_H;
    lstm_pack_w<<<(n1 + 255) / 256, 256, 0, stream>>>(W_hh0, whh0_pk, LSTM_H);
    lstm_pack_w<<<(n1 + 255) / 256, 256, 0, stream>>>(W_ih1, wih1_pk, LSTM_H);
    lstm_pack_w<<<(n1 + 255) / 256, 256, 0, stream>>>(W_hh1, whh1_pk, LSTM_H);
  }

  // Layer 0: persistent recurrence, writes f16 h1 trace.
  lstm_rec<0><<<LSTM_B / 16, 512, 0, stream>>>(
      x, nullptr, wih0_pk, whh0_pk, b_ih0, b_hh0, h1, nullptr, nullptr, nullptr);

  // Layer 1: persistent recurrence + FC head.
  lstm_rec<1><<<LSTM_B / 16, 512, 0, stream>>>(
      nullptr, h1, wih1_pk, whh1_pk, b_ih1, b_hh1, nullptr, fc_w, fc_b, out);
}
